// NeuralCDE_61435212202254
// MI455X (gfx1250) — matrix operand order repaired and checked by reading, not compiled, not run on hardware
//
#include <hip/hip_runtime.h>

// ---------------- problem constants (match reference) ----------------
#define BQ     4096      // batch
#define NSEG   63        // SEQ-1
#define DD     20        // D
#define HD     256       // H == W == 256 (z width)
#define N3     5120      // H*D = W3 output width
#define NSTEPS 126
#define DTC    0.5f
// ---------------- kernel tiling ----------------
#define MT     32        // batch rows per workgroup
#define LDA    264       // LDS row stride in bf16 halves (132 dwords -> 4-bank step)
#define NTHR   256       // 8 waves of 32

typedef __attribute__((ext_vector_type(16))) __bf16        v16bf;
typedef __attribute__((ext_vector_type(8)))  float         v8f;
typedef __attribute__((ext_vector_type(4)))  unsigned int  v4u;

union Frag { v16bf bf; v4u u[2]; };

__device__ __forceinline__ unsigned short f2bf(float f) {
  unsigned int u = __float_as_uint(f);
  u += 0x7FFFu + ((u >> 16) & 1u);           // round-to-nearest-even
  return (unsigned short)(u >> 16);
}
// branch-free fast tanh: 1 - 2/(e^{2x}+1)  (v_exp + v_rcp, exact at +-inf)
__device__ __forceinline__ float ftanh(float x) {
  float e = __expf(2.f * x);
  return 1.f - 2.f * __builtin_amdgcn_rcpf(e + 1.f);
}
__device__ __forceinline__ float sigm(float x) { return 1.f / (1.f + __expf(-x)); }
__device__ __forceinline__ float sofp(float x) { return (x > 20.f) ? x : log1pf(__expf(x)); }

// ---- prep kernel: f32 weights -> bf16 in workspace --------------------------
// layout in d_ws (ushort): W1[65536] | W2[65536] | W3[1310720] | Ws[65536]
__global__ void __launch_bounds__(256) cvt_bf16_kernel(
    const float* __restrict__ W1, const float* __restrict__ W2,
    const float* __restrict__ W3, const float* __restrict__ Ws,
    unsigned short* __restrict__ dst) {
  for (int i = blockIdx.x * blockDim.x + threadIdx.x; i < 1507328;
       i += gridDim.x * blockDim.x) {
    float v;
    if      (i < 65536)   v = W1[i];
    else if (i < 131072)  v = W2[i - 65536];
    else if (i < 1441792) v = W3[i - 131072];
    else                  v = Ws[i - 1441792];
    dst[i] = f2bf(v);
  }
}

// ---- persistent Neural-CDE kernel: one WG integrates 32 batch rows ---------
__global__ void __launch_bounds__(NTHR) ncde_persistent_kernel(
    const float* __restrict__ coeffs,
    const float* __restrict__ Wi,   const float* __restrict__ bi,
    const float* __restrict__ b1,   const float* __restrict__ b2,
    const float* __restrict__ b3,   const float* __restrict__ bs,
    const float* __restrict__ lng,  const float* __restrict__ lnb,
    const float* __restrict__ Winf, const float* __restrict__ binf,
    const float* __restrict__ Wdet, const float* __restrict__ bdet,
    const float* __restrict__ Wsur, const float* __restrict__ bsur,
    const unsigned short* __restrict__ W1b, const unsigned short* __restrict__ W2b,
    const unsigned short* __restrict__ W3b, const unsigned short* __restrict__ Wsb,
    float* __restrict__ out)
{
  __shared__ __align__(16) float          zbs[MT * HD];     // z state (f32)
  __shared__ __align__(16) float          kcs[MT * HD];     // RK4 weighted-k accum
  __shared__ __align__(16) float          vfs[MT * HD];     // k_s accumulator (f32)
  __shared__ __align__(16) unsigned short actA[MT * LDA];   // bf16 activations ping
  __shared__ __align__(16) unsigned short actB[MT * LDA];   // bf16 activations pong
  __shared__ __align__(16) float          dXs[MT * DD];     // spline derivative tile
  __shared__ float mus[MT];
  __shared__ float rss[MT];

  const int tid  = threadIdx.x;
  const int lane = tid & 31;
  const int wv   = tid >> 5;       // wave id 0..7
  const int hlf  = lane >> 4;      // 0/1: lane half (A: K offset, C: M offset)
  const int lc   = lane & 15;
  const int m0   = blockIdx.x * MT;

  // ---- init: z0 = coeffs[:,0,:D] @ Wi + bi  (thread owns column tid, all rows)
  {
    float wcol[DD];
    #pragma unroll
    for (int d = 0; d < DD; ++d) wcol[d] = Wi[d * HD + tid];
    float bic = bi[tid];
    #pragma unroll
    for (int r = 0; r < MT; ++r) {
      const float* cr = coeffs + (size_t)(m0 + r) * (NSEG * 4 * DD); // segment 0
      float acc = bic;
      #pragma unroll
      for (int d = 0; d < DD; ++d) acc += cr[d] * wcol[d];
      zbs[r * HD + tid] = acc;
      kcs[r * HD + tid] = 0.f;
      vfs[r * HD + tid] = 0.f;
    }
  }
  __syncthreads();

  // ================= RK4 time loop =================
  for (int step = 0; step < NSTEPS; ++step) {
    const float t0 = (float)step * DTC;
    for (int s = 0; s < 4; ++s) {
      const float cin = (s == 0) ? 0.f : ((s == 3) ? DTC : 0.5f * DTC);
      const float wk  = (s == 3) ? 2.f : (float)s;   // fold weight of k_{s-1}

      // ---- prep: stage input -> actA (bf16); fold k_{s-1}; zero vfs
      for (int i = tid; i < MT * HD; i += NTHR) {
        float kprev = vfs[i];
        kcs[i] += wk * kprev;
        int r = i >> 8, c = i & 255;
        actA[r * LDA + c] = f2bf(zbs[i] + cin * kprev);
        vfs[i] = 0.f;
      }
      // ---- dXdt at this stage time (scalar t, per-row coefficients)
      {
        float t = t0 + cin;          // stage times: t0, t0+.25, t0+.25, t0+.5
        int iseg = (int)t;
        if (iseg > NSEG - 1) iseg = NSEG - 1;
        float u = t - (float)iseg;
        for (int i = tid; i < MT * DD; i += NTHR) {
          int r = i / DD, d = i - r * DD;
          const float* cr = coeffs + ((size_t)(m0 + r) * NSEG + iseg) * (4 * DD);
          dXs[i] = cr[DD + d] + 2.f * cr[2*DD + d] * u + 3.f * cr[3*DD + d] * (u * u);
        }
      }
      __syncthreads();

      // ---- GEMM1: actB = tanh(actA @ W1 + b1)   (32x256 @ 256x256)
      // wave wv: row-tile mt = wv>>2, n-tiles (wv&3)*4 .. +3.
      // A-frags (8 k-slabs) hoisted to registers once; inner loop is B-load + WMMA.
      {
        const int mt = wv >> 2;
        const int mrow = mt * 16 + lc;
        Frag fas[8];
        #pragma unroll
        for (int i = 0; i < 8; ++i) {
          fas[i].u[0] = *(const v4u*)&actA[mrow * LDA + i * 32 + hlf * 8];
          fas[i].u[1] = *(const v4u*)&actA[mrow * LDA + i * 32 + hlf * 8 + 16];
        }
        #pragma unroll 1
        for (int j = 0; j < 4; ++j) {
          int n0 = ((wv & 3) * 4 + j) * 16;
          v8f c8 = {};
          #pragma unroll
          for (int i = 0; i < 8; ++i) {
            Frag fb;
            const unsigned short* pb = W1b + (size_t)(i * 32 + hlf * 8) * HD + n0 + lc;
            unsigned short* fe = (unsigned short*)&fb;
            for (int s = 0; s < 8; ++s) { fe[s] = pb[s * HD]; fe[s + 8] = pb[(s + 16) * HD]; }
            c8 = __builtin_amdgcn_wmma_f32_16x16x32_bf16(
                     false, fas[i].bf, false, fb.bf, (short)0, c8, false, false);
          }
          float bias = b1[n0 + lc];
          #pragma unroll
          for (int r = 0; r < 8; ++r)
            actB[(mt*16 + r + 8*hlf) * LDA + n0 + lc] = f2bf(ftanh(c8[r] + bias));
        }
      }
      __syncthreads();

      // ---- GEMM2: actA = tanh(actB @ W2 + b2)
      {
        const int mt = wv >> 2;
        const int mrow = mt * 16 + lc;
        Frag fas[8];
        #pragma unroll
        for (int i = 0; i < 8; ++i) {
          fas[i].u[0] = *(const v4u*)&actB[mrow * LDA + i * 32 + hlf * 8];
          fas[i].u[1] = *(const v4u*)&actB[mrow * LDA + i * 32 + hlf * 8 + 16];
        }
        #pragma unroll 1
        for (int j = 0; j < 4; ++j) {
          int n0 = ((wv & 3) * 4 + j) * 16;
          v8f c8 = {};
          #pragma unroll
          for (int i = 0; i < 8; ++i) {
            Frag fb;
            const unsigned short* pb = W2b + (size_t)(i * 32 + hlf * 8) * HD + n0 + lc;
            unsigned short* fe = (unsigned short*)&fb;
            for (int s = 0; s < 8; ++s) { fe[s] = pb[s * HD]; fe[s + 8] = pb[(s + 16) * HD]; }
            c8 = __builtin_amdgcn_wmma_f32_16x16x32_bf16(
                     false, fas[i].bf, false, fb.bf, (short)0, c8, false, false);
          }
          float bias = b2[n0 + lc];
          #pragma unroll
          for (int r = 0; r < 8; ++r)
            actA[(mt*16 + r + 8*hlf) * LDA + n0 + lc] = f2bf(ftanh(c8[r] + bias));
        }
      }
      __syncthreads();

      // ---- GEMM3 + einsum fold: vf[b,h] += (actA @ W3)[b, h*20+d] * dX[b,d]
      // wave wv owns columns [wv*640, wv*640+640) -> h in [wv*32, wv*32+32).
      // Both 16-row A tiles (all 8 k-slabs) hoisted to registers (128 VGPRs);
      // inner loop is a pure B-stream (global/L2) + 2 WMMAs per slab.
      {
        Frag fa0s[8], fa1s[8];
        #pragma unroll
        for (int i = 0; i < 8; ++i) {
          fa0s[i].u[0] = *(const v4u*)&actA[lc * LDA + i * 32 + hlf * 8];
          fa0s[i].u[1] = *(const v4u*)&actA[lc * LDA + i * 32 + hlf * 8 + 16];
          fa1s[i].u[0] = *(const v4u*)&actA[(16 + lc) * LDA + i * 32 + hlf * 8];
          fa1s[i].u[1] = *(const v4u*)&actA[(16 + lc) * LDA + i * 32 + hlf * 8 + 16];
        }
        const int nbase = wv * 640;
        #pragma unroll 1
        for (int jn = 0; jn < 40; ++jn) {
          int n0 = nbase + jn * 16;
          // prefetch next tile's B columns (global_prefetch_b8; W3 stream = bulk of bytes)
          if (jn + 1 < 40)
            __builtin_prefetch(&W3b[(size_t)(lane * 8) * N3 + n0 + 16], 0, 1);
          int colg = n0 + lc;
          int hidx = colg / DD;
          int didx = colg - hidx * DD;
          v8f c0 = {}, c1 = {};
          #pragma unroll
          for (int i = 0; i < 8; ++i) {
            Frag fb;
            const unsigned short* pb = W3b + (size_t)(i * 32 + hlf * 8) * N3 + n0 + lc;
            unsigned short* fe = (unsigned short*)&fb;
            for (int s = 0; s < 8; ++s) { fe[s] = pb[(size_t)s * N3]; fe[s + 8] = pb[(size_t)(s + 16) * N3]; }
            c0 = __builtin_amdgcn_wmma_f32_16x16x32_bf16(
                     false, fa0s[i].bf, false, fb.bf, (short)0, c0, false, false);
            c1 = __builtin_amdgcn_wmma_f32_16x16x32_bf16(
                     false, fa1s[i].bf, false, fb.bf, (short)0, c1, false, false);
          }
          #pragma unroll
          for (int r = 0; r < 8; ++r) {
            int r0 = r + 8 * hlf;
            atomicAdd(&vfs[r0 * HD + hidx], c0[r] * dXs[r0 * DD + didx]);
            int r1 = 16 + r + 8 * hlf;
            atomicAdd(&vfs[r1 * HD + hidx], c1[r] * dXs[r1 * DD + didx]);
          }
        }
      }
      __syncthreads();

      // ---- b3 contribution: vf[b,h] += sum_d b3[h*20+d]*dX[b,d]
      // thread column h = tid fixed -> b3 row hoisted by compiler (L2 resident)
      {
        const float* bp = b3 + tid * DD;
        #pragma unroll
        for (int r = 0; r < MT; ++r) {
          float acc = 0.f;
          #pragma unroll
          for (int d = 0; d < DD; ++d) acc += bp[d] * dXs[r * DD + d];
          vfs[r * HD + tid] += acc;
        }
      }
      __syncthreads();
    } // stages

    // ---- z update: z += DT/6 * (k1 + 2k2 + 2k3 + k4)
    for (int i = tid; i < MT * HD; i += NTHR) {
      zbs[i] += (DTC / 6.f) * (kcs[i] + vfs[i]);
      kcs[i] = 0.f;
      vfs[i] = 0.f;
    }
    __syncthreads();
  } // steps

  // ================= head =================
  // h = z @ Ws + bs -> vfs (f32)
  for (int i = tid; i < MT * HD; i += NTHR) {
    int r = i >> 8, c = i & 255;
    actA[r * LDA + c] = f2bf(zbs[i]);
  }
  __syncthreads();
  {
    const int mt = wv >> 2;
    const int mrow = mt * 16 + lc;
    Frag fas[8];
    #pragma unroll
    for (int i = 0; i < 8; ++i) {
      fas[i].u[0] = *(const v4u*)&actA[mrow * LDA + i * 32 + hlf * 8];
      fas[i].u[1] = *(const v4u*)&actA[mrow * LDA + i * 32 + hlf * 8 + 16];
    }
    #pragma unroll 1
    for (int j = 0; j < 4; ++j) {
      int n0 = ((wv & 3) * 4 + j) * 16;
      v8f c8 = {};
      #pragma unroll
      for (int i = 0; i < 8; ++i) {
        Frag fb;
        const unsigned short* pb = Wsb + (size_t)(i * 32 + hlf * 8) * HD + n0 + lc;
        unsigned short* fe = (unsigned short*)&fb;
        for (int s = 0; s < 8; ++s) { fe[s] = pb[s * HD]; fe[s + 8] = pb[(s + 16) * HD]; }
        c8 = __builtin_amdgcn_wmma_f32_16x16x32_bf16(
                 false, fas[i].bf, false, fb.bf, (short)0, c8, false, false);
      }
      float bias = bs[n0 + lc];
      #pragma unroll
      for (int r = 0; r < 8; ++r)
        vfs[(mt*16 + r + 8*hlf) * HD + n0 + lc] = c8[r] + bias;
    }
  }
  __syncthreads();
  // LayerNorm stats (one thread per row)
  if (tid < MT) {
    float s1 = 0.f, s2 = 0.f;
    for (int c = 0; c < HD; ++c) { float v = vfs[tid * HD + c]; s1 += v; s2 += v * v; }
    float mu = s1 * (1.f / HD);
    mus[tid] = mu;
    rss[tid] = rsqrtf(s2 * (1.f / HD) - mu * mu + 1e-5f);
  }
  __syncthreads();
  // r = relu(g*(h-mu)*rsqrt(var+eps)+b) in place; also write r output
  for (int i = tid; i < MT * HD; i += NTHR) {
    int r = i >> 8, c = i & 255;
    float v = lng[c] * (vfs[i] - mus[r]) * rss[r] + lnb[c];
    v = fmaxf(v, 0.f);
    vfs[i] = v;
    out[(size_t)(BQ * 9) + (size_t)(m0 + r) * HD + c] = v;
  }
  __syncthreads();
  // tiny heads: sigmoid(r@Winf+binf), sigmoid(r@Wdet+bdet), softplus(r@Wsur+bsur)
  for (int i = tid; i < MT * 9; i += NTHR) {
    int r = i / 9, o = i - r * 9;
    int b = m0 + r;
    float dot = 0.f;
    if (o < 4) {
      for (int c = 0; c < HD; ++c) dot += vfs[r * HD + c] * Winf[c * 4 + o];
      out[(size_t)b * 4 + o] = sigm(dot + binf[o]);
    } else if (o < 7) {
      int oo = o - 4;
      for (int c = 0; c < HD; ++c) dot += vfs[r * HD + c] * Wdet[c * 3 + oo];
      out[(size_t)(BQ * 4) + (size_t)b * 3 + oo] = sigm(dot + bdet[oo]);
    } else {
      int oo = o - 7;
      for (int c = 0; c < HD; ++c) dot += vfs[r * HD + c] * Wsur[c * 2 + oo];
      out[(size_t)(BQ * 7) + (size_t)b * 2 + oo] = sofp(dot + bsur[oo]);
    }
  }
}

extern "C" void kernel_launch(void* const* d_in, const int* in_sizes, int n_in,
                              void* d_out, int out_size, void* d_ws, size_t ws_size,
                              hipStream_t stream) {
  const float* coeffs = (const float*)d_in[0];
  const float* Wi   = (const float*)d_in[1];
  const float* bi   = (const float*)d_in[2];
  const float* W1   = (const float*)d_in[3];
  const float* b1   = (const float*)d_in[4];
  const float* W2   = (const float*)d_in[5];
  const float* b2   = (const float*)d_in[6];
  const float* W3   = (const float*)d_in[7];
  const float* b3   = (const float*)d_in[8];
  const float* Ws   = (const float*)d_in[9];
  const float* bs   = (const float*)d_in[10];
  const float* lng  = (const float*)d_in[11];
  const float* lnb  = (const float*)d_in[12];
  const float* Winf = (const float*)d_in[13];
  const float* binf = (const float*)d_in[14];
  const float* Wdet = (const float*)d_in[15];
  const float* bdet = (const float*)d_in[16];
  const float* Wsur = (const float*)d_in[17];
  const float* bsur = (const float*)d_in[18];

  unsigned short* wb = (unsigned short*)d_ws;   // bf16 weight cache

  cvt_bf16_kernel<<<512, 256, 0, stream>>>(W1, W2, W3, Ws, wb);

  ncde_persistent_kernel<<<BQ / MT, NTHR, 0, stream>>>(
      coeffs, Wi, bi, b1, b2, b3, bs, lng, lnb,
      Winf, binf, Wdet, bdet, Wsur, bsur,
      wb, wb + 65536, wb + 131072, wb + 1441792,
      (float*)d_out);

  (void)in_sizes; (void)n_in; (void)out_size; (void)ws_size;
}
